// FairNTXent_61933428408649
// MI455X (gfx1250) — matrix operand order repaired and checked by reading, not compiled, not run on hardware
//
#include <hip/hip_runtime.h>
#include <math.h>

// ---------------------------------------------------------------------------
// FairNTXent on MI455X (gfx1250, wave32, WMMA).
//
// Heavy term: den[i] = sum_j exp( (zn_i . zn_j) / T )  over 8192x8192, done
// with v_wmma_f32_16x16x32_bf16 (f32 accumulate), exp fused per tile (native
// v_exp_f32 via __expf), no S matrix ever stored. Fairness term reduced
// algebraically to |sum_{i in g} zn_i|^2.
// ---------------------------------------------------------------------------

typedef __bf16 bf16_t;
typedef __attribute__((ext_vector_type(8)))  bf16_t bf16x8;
typedef __attribute__((ext_vector_type(16))) bf16_t bf16x16;
typedef __attribute__((ext_vector_type(8)))  float  f32x8;

#define BB   4096          // batch
#define DD   256           // feature dim
#define N2   8192          // 2*B rows of Z = [z_i; z_j]

#define TEMP_INV    2.0f                   // 1/T, T = 0.5
#define LN2_F       0.6931471805599453f

// -------------------- small reduction helper (wave32) ----------------------
__device__ __forceinline__ float waveSum(float v) {
#pragma unroll
  for (int m = 16; m >= 1; m >>= 1) v += __shfl_xor(v, m, 32);
  return v;
}

// -------------------- K1: normalize rows, emit bf16 Z, zero scratch --------
__global__ void __launch_bounds__(256) k_normalize(
    const float* __restrict__ zi, const float* __restrict__ zj,
    bf16_t* __restrict__ ZB, float* __restrict__ invn,
    float* __restrict__ qii, float* __restrict__ rowsum,
    float* __restrict__ Vsum) {
  const int r = blockIdx.x;      // row of concatenated Z (0..8191)
  const int t = threadIdx.x;     // feature index (0..255)
  const float* src = (r < BB) ? (zi + (size_t)r * DD) : (zj + (size_t)(r - BB) * DD);
  const float v = src[t];

  __shared__ float sb[9];
  float w = waveSum(v * v);
  if ((t & 31) == 0) sb[t >> 5] = w;
  __syncthreads();
  if (t == 0) {
    float s = 0.f;
    for (int i = 0; i < 8; i++) s += sb[i];
    sb[8] = s;
  }
  __syncthreads();
  const float inv = 1.0f / fmaxf(sqrtf(sb[8]), 1e-8f);

  const bf16_t b = (bf16_t)(v * inv);
  ZB[(size_t)r * DD + t] = b;
  const float fb = (float)b;

  __syncthreads();                       // sb reuse
  float w2 = waveSum(fb * fb);           // diagonal S_ii as the WMMA will see it
  if ((t & 31) == 0) sb[t >> 5] = w2;
  __syncthreads();
  if (t == 0) {
    float s = 0.f;
    for (int i = 0; i < 8; i++) s += sb[i];
    qii[r]    = s;
    invn[r]   = inv;
    rowsum[r] = 0.0f;                    // zero scratch every launch
  }
  if (r < 4) Vsum[r * DD + t] = 0.0f;    // zero the 4 group vectors
}

// -------------------- K2: pos[i] = zn_i . zn_j (fp32 exact) ----------------
__global__ void __launch_bounds__(256) k_pos(
    const float* __restrict__ zi, const float* __restrict__ zj,
    const float* __restrict__ invn, float* __restrict__ pos) {
  const int r = blockIdx.x, t = threadIdx.x;
  float v = zi[(size_t)r * DD + t] * zj[(size_t)r * DD + t];
  __shared__ float sb[8];
  float w = waveSum(v);
  if ((t & 31) == 0) sb[t >> 5] = w;
  __syncthreads();
  if (t == 0) {
    float s = 0.f;
    for (int i = 0; i < 8; i++) s += sb[i];
    pos[r] = s * invn[r] * invn[r + BB];
  }
}

// -------------------- K3: group vectors v_g = sum_{i in g} zn_i ------------
__global__ void __launch_bounds__(256) k_groupvec(
    const float* __restrict__ zi, const int* __restrict__ sf,
    const float* __restrict__ invn, float* __restrict__ Vsum) {
  const int d = threadIdx.x;             // feature dim
  const int b = blockIdx.x;              // 8 blocks x 512 rows each
  float a0 = 0.f, a1 = 0.f, a2 = 0.f, a3 = 0.f;
  const int r0 = b * (BB / 8), r1 = r0 + (BB / 8);
  for (int r = r0; r < r1; r++) {
    const float v = zi[(size_t)r * DD + d] * invn[r];
    const int g = sf[r];
    a0 += (g == 0) ? v : 0.f;
    a1 += (g == 1) ? v : 0.f;
    a2 += (g == 2) ? v : 0.f;
    a3 += (g == 3) ? v : 0.f;
  }
  atomicAdd(&Vsum[0 * DD + d], a0);
  atomicAdd(&Vsum[1 * DD + d], a1);
  atomicAdd(&Vsum[2 * DD + d], a2);
  atomicAdd(&Vsum[3 * DD + d], a3);
}

// -------------------- K4: WMMA exp-rowsum over full 8192x8192 --------------
#define ROWS_PER_BLOCK 128             // 8 waves x 16 rows
#define COLCHUNKS      4               // column split for occupancy
#define CHUNK_COLS     (N2 / COLCHUNKS)      // 2048
#define STAGE_COLS     32              // columns staged in LDS per step
#define STAGES         (CHUNK_COLS / STAGE_COLS)   // 64
#define LDS_STRIDE     132             // dwords per staged column (128 data + 4 pad)

__global__ void __launch_bounds__(256, 1) k_simrow(
    const bf16_t* __restrict__ ZB, float* __restrict__ rowsum) {
  __shared__ uint32_t ldsB[STAGE_COLS * LDS_STRIDE];  // 32 cols x 256 K bf16 (+pad)

  const int t    = threadIdx.x;
  const int lane = t & 31;
  const int w    = t >> 5;
  const int rb   = blockIdx.x / COLCHUNKS;
  const int cc   = blockIdx.x % COLCHUNKS;
  const int i0   = rb * ROWS_PER_BLOCK + w * 16;   // this wave's 16 rows
  const int hsel = lane >> 4;                      // half-wave selector
  const int mrow = lane & 15;                      // row/col within tile

  // A fragments: 16 rows x K=256 resident in registers (8 x v16bf16).
  // A layout (matches B): lane<16 holds K[k0..k0+15];
  // lane>=16 holds K[k0+16..k0+31].
  bf16x16 A[8];
  const bf16_t* arow = ZB + (size_t)(i0 + mrow) * DD;
#pragma unroll
  for (int kk = 0; kk < 8; kk++) {
    const int k0 = kk * 32;
    bf16x8 lo = *reinterpret_cast<const bf16x8*>(arow + k0 + hsel * 16);
    bf16x8 hi = *reinterpret_cast<const bf16x8*>(arow + k0 + hsel * 16 + 8);
    A[kk] = __builtin_shufflevector(lo, hi, 0, 1, 2, 3, 4, 5, 6, 7,
                                    8, 9, 10, 11, 12, 13, 14, 15);
  }

  float acc[8];
#pragma unroll
  for (int r = 0; r < 8; r++) acc[r] = 0.f;

  // cooperative staging: 32 cols x 512 B = 16 KB -> 64 B per thread
  const int ldcol = t >> 3;   // 0..31
  const int ldseg = t & 7;    // 64-byte segment

  for (int s = 0; s < STAGES; s++) {
    const int j0 = cc * CHUNK_COLS + s * STAGE_COLS;
    __syncthreads();  // previous stage fully consumed
    {
      const uint4* src = reinterpret_cast<const uint4*>(
          ZB + (size_t)(j0 + ldcol) * DD + ldseg * 32);
      uint4* dst = reinterpret_cast<uint4*>(&ldsB[ldcol * LDS_STRIDE + ldseg * 16]);
      dst[0] = src[0];
      dst[1] = src[1];
      dst[2] = src[2];
      dst[3] = src[3];
    }
    __syncthreads();  // stage visible to all waves

    // two independent accumulator chains -> WMMA pipe stays fed
    f32x8 c0 = {};
    f32x8 c1 = {};
#pragma unroll
    for (int kk = 0; kk < 8; kk++) {
      // B (KxN) layout: lane holds column lane%16, 16 contiguous K values
      // starting at k0 + 16*(lane/16), packed 2 bf16 per VGPR.
      union { uint4 u[2]; bf16x16 v; } b0, b1;
      const uint4* p0 = reinterpret_cast<const uint4*>(
          &ldsB[mrow * LDS_STRIDE + kk * 16 + hsel * 8]);
      const uint4* p1 = reinterpret_cast<const uint4*>(
          &ldsB[(16 + mrow) * LDS_STRIDE + kk * 16 + hsel * 8]);
      b0.u[0] = p0[0]; b0.u[1] = p0[1];
      b1.u[0] = p1[0]; b1.u[1] = p1[1];
      c0 = __builtin_amdgcn_wmma_f32_16x16x32_bf16(
          false, A[kk], false, b0.v, (short)0, c0, false, false);
      c1 = __builtin_amdgcn_wmma_f32_16x16x32_bf16(
          false, A[kk], false, b1.v, (short)0, c1, false, false);
    }
#pragma unroll
    for (int r = 0; r < 8; r++)
      acc[r] += __expf(c0[r] * TEMP_INV) + __expf(c1[r] * TEMP_INV);
  }

  // C layout: VGPR r, lanes 0-15 -> row i0+r ; lanes 16-31 -> row i0+r+8.
#pragma unroll
  for (int r = 0; r < 8; r++) {
    float a = acc[r];
    a += __shfl_xor(a, 1, 32);
    a += __shfl_xor(a, 2, 32);
    a += __shfl_xor(a, 4, 32);
    a += __shfl_xor(a, 8, 32);
    if ((lane & 15) == 0) atomicAdd(&rowsum[i0 + r + hsel * 8], a);
  }
}

// -------------------- K5: scalar finalize ----------------------------------
__global__ void __launch_bounds__(256) k_final(
    const int* __restrict__ sf, const float* __restrict__ rowsum,
    const float* __restrict__ qii, const float* __restrict__ pos,
    const float* __restrict__ Vsum, float* __restrict__ out) {
  __shared__ int   scnt[4];
  __shared__ float sb[9];
  __shared__ float sgs[4];
  const int t = threadIdx.x;

  if (t < 4) scnt[t] = 0;
  __syncthreads();
  int c0 = 0, c1 = 0, c2 = 0, c3 = 0;
  for (int r = t; r < BB; r += 256) {
    const int g = sf[r];
    c0 += (g == 0); c1 += (g == 1); c2 += (g == 2); c3 += (g == 3);
  }
  atomicAdd(&scnt[0], c0); atomicAdd(&scnt[1], c1);
  atomicAdd(&scnt[2], c2); atomicAdd(&scnt[3], c3);
  __syncthreads();

  // contrastive = mean over 2B rows of ( log(den) - pos/T )
  float part = 0.f;
  for (int r = t; r < N2; r += 256) {
    const float den = rowsum[r] - __expf(qii[r] * TEMP_INV);  // drop diagonal
    const float p   = pos[r & (BB - 1)];
    part += log2f(den) * LN2_F - p * TEMP_INV;
  }
  float w = waveSum(part);
  if ((t & 31) == 0) sb[t >> 5] = w;
  __syncthreads();
  if (t == 0) {
    float s = 0.f;
    for (int i = 0; i < 8; i++) s += sb[i];
    sb[8] = s;
  }
  __syncthreads();
  const float contrastive = sb[8] / (float)N2;

  // group_sums[g] = |v_g|^2
  for (int g = 0; g < 4; g++) {
    const float v = Vsum[g * DD + t];
    float ws = waveSum(v * v);
    __syncthreads();
    if ((t & 31) == 0) sb[t >> 5] = ws;
    __syncthreads();
    if (t == 0) {
      float s = 0.f;
      for (int i = 0; i < 8; i++) s += sb[i];
      sgs[g] = s;
    }
    __syncthreads();
  }

  if (t == 0) {
    float fsum = 0.f, nu = 0.f;
    for (int g = 0; g < 4; g++) {
      const float c = (float)scnt[g];
      if (c > 0.5f) nu += 1.0f;
      if (c > 1.5f) fsum += sgs[g] / (c * (c - 1.0f));
    }
    const float fairness = 0.1f * (fsum / nu);
    out[0] = contrastive + fairness;
  }
}

// -------------------- launcher ---------------------------------------------
extern "C" void kernel_launch(void* const* d_in, const int* in_sizes, int n_in,
                              void* d_out, int out_size, void* d_ws, size_t ws_size,
                              hipStream_t stream) {
  const float* zi = (const float*)d_in[0];
  const float* zj = (const float*)d_in[1];
  const int*   sf = (const int*)d_in[2];

  char* ws = (char*)d_ws;
  bf16_t* ZB   = (bf16_t*)ws;
  size_t off = (size_t)N2 * DD * sizeof(bf16_t);        // 4 MB
  float* invn   = (float*)(ws + off); off += (size_t)N2 * 4;
  float* qii    = (float*)(ws + off); off += (size_t)N2 * 4;
  float* pos    = (float*)(ws + off); off += (size_t)BB * 4;
  float* rowsum = (float*)(ws + off); off += (size_t)N2 * 4;
  float* Vsum   = (float*)(ws + off); off += (size_t)4 * DD * 4;

  k_normalize<<<N2, 256, 0, stream>>>(zi, zj, ZB, invn, qii, rowsum, Vsum);
  k_pos<<<BB, 256, 0, stream>>>(zi, zj, invn, pos);
  k_groupvec<<<8, 256, 0, stream>>>(zi, sf, invn, Vsum);
  k_simrow<<<(N2 / ROWS_PER_BLOCK) * COLCHUNKS, 256, 0, stream>>>(ZB, rowsum);
  k_final<<<1, 256, 0, stream>>>(sf, rowsum, qii, pos, Vsum, (float*)d_out);
}
